// A2GNN_47510928228592
// MI455X (gfx1250) — matrix operand order repaired and checked by reading, not compiled, not run on hardware
//
#include <hip/hip_runtime.h>

typedef __attribute__((ext_vector_type(16))) _Float16 v16h;
typedef __attribute__((ext_vector_type(8)))  _Float16 v8h;
typedef __attribute__((ext_vector_type(8)))  float    v8f;
typedef __attribute__((ext_vector_type(4)))  float    v4f;

#define N_      8192
#define NFEAT_  512
#define NHID_   128
#define NCLASS_ 21
#define NW_     256   /* 8192/32 mask words per row */

static __device__ __forceinline__ v16h cat16(v8h lo, v8h hi) {
  v16h r;
#pragma unroll
  for (int i = 0; i < 8; ++i) { r[i] = lo[i]; r[8 + i] = hi[i]; }
  return r;
}

static __device__ __forceinline__ v8f wmma_f16(v16h a, v16h b, v8f c) {
  // D = A(16x32 f16) * B(32x16 f16) + C(16x16 f32)
  return __builtin_amdgcn_wmma_f32_16x16x32_f16(false, a, false, b, (short)0, c,
                                                false, false);
}

// ---------------------------------------------------------------------------
// Pack adj/aff (0/1-valued f32, 256MB each) into bitmasks (8MB each -> L2).
// One block per row; each wave ballots 32 consecutive j's into one word.
// ---------------------------------------------------------------------------
__global__ __launch_bounds__(256) void pack_kernel(const float* __restrict__ adj,
                                                   const float* __restrict__ aff,
                                                   unsigned* __restrict__ adjw,
                                                   unsigned* __restrict__ affw) {
  const int i = blockIdx.x;
  const int wv = threadIdx.x >> 5;
  const int lane = threadIdx.x & 31;
  for (int it = 0; it < 32; ++it) {
    const int w = it * 8 + wv;
    const int j = w * 32 + lane;
    const size_t idx = (size_t)i * N_ + j;
    const float a = adj[idx];
    const float f = aff[idx];
    const unsigned am = (unsigned)__ballot(a != 0.f);
    const unsigned fm = (unsigned)__ballot(f != 0.f);
    if (lane == 0) { adjw[i * NW_ + w] = am; affw[i * NW_ + w] = fm; }
  }
}

// ---------------------------------------------------------------------------
// Transpose + convert w_embed [512,128] f32 -> wT16 [128][512] f16
// ---------------------------------------------------------------------------
__global__ __launch_bounds__(256) void wt_kernel(const float* __restrict__ w,
                                                 _Float16* __restrict__ wT) {
  const int idx = blockIdx.x * 256 + threadIdx.x;  // 65536 elements
  const int k = idx >> 7;
  const int n = idx & 127;
  wT[(size_t)n * NFEAT_ + k] = (_Float16)w[idx];
}

// ---------------------------------------------------------------------------
// h = relu(x @ w_embed) via WMMA.
// Each wave: one 16-row x 64-col tile (4 WMMA column tiles), K=512.
// 1024 waves total (2x the old tiling) for latency hiding at low occupancy.
// ---------------------------------------------------------------------------
__global__ __launch_bounds__(256) void embed_kernel(const float* __restrict__ x,
                                                    const _Float16* __restrict__ wT,
                                                    float* __restrict__ h) {
  const int wv = threadIdx.x >> 5;
  const int lane = threadIdx.x & 31;
  const int hf = lane >> 4;     // lane half (K-split in A/B layouts)
  const int lm = lane & 15;     // M (A) / N (B) index
  const int g = blockIdx.x * 8 + wv;      // 0..1023
  const int row0 = (g >> 1) * 16;         // 16-row tile
  const int t0 = (g & 1) * 4;             // column-tile group: tiles t0..t0+3

  v8f acc[4];
#pragma unroll
  for (int t = 0; t < 4; ++t) acc[t] = {};

  const float* xr = x + (size_t)(row0 + lm) * NFEAT_;
  for (int kc = 0; kc < 16; ++kc) {           // K chunks of 32
    // A: x rows, f32 -> f16.  lanes<16: K {0..7,16..23}; lanes>=16: {8..15,24..31}
    const v4f f0 = *(const v4f*)(xr + kc * 32 + hf * 8);
    const v4f f1 = *(const v4f*)(xr + kc * 32 + hf * 8 + 4);
    const v4f f2 = *(const v4f*)(xr + kc * 32 + 16 + hf * 8);
    const v4f f3 = *(const v4f*)(xr + kc * 32 + 16 + hf * 8 + 4);
    // B: issue the 4 column tiles' loads first, then the 4 WMMAs.
    v8h blo[4], bhi[4];
#pragma unroll
    for (int t = 0; t < 4; ++t) {
      const _Float16* wr =
          wT + (size_t)((t0 + t) * 16 + lm) * NFEAT_ + kc * 32 + hf * 8;
      blo[t] = *(const v8h*)wr;
      bhi[t] = *(const v8h*)(wr + 16);
    }
    v16h a;
#pragma unroll
    for (int e = 0; e < 4; ++e) {
      a[e]      = (_Float16)f0[e];
      a[4 + e]  = (_Float16)f1[e];
      a[8 + e]  = (_Float16)f2[e];
      a[12 + e] = (_Float16)f3[e];
    }
#pragma unroll
    for (int t = 0; t < 4; ++t)
      acc[t] = wmma_f16(a, cat16(blo[t], bhi[t]), acc[t]);
  }
  // C layout: vgpr r, lanes<16: M=r,N=lane ; lanes>=16: M=r+8,N=lane-16
#pragma unroll
  for (int t = 0; t < 4; ++t)
#pragma unroll
    for (int r = 0; r < 8; ++r) {
      const int m = r + 8 * hf;
      h[(size_t)(row0 + m) * NHID_ + (t0 + t) * 16 + lm] = fmaxf(acc[t][r], 0.f);
    }
}

// ---------------------------------------------------------------------------
// Per-row prep: hn16 = f16 row-normalized h (for cosine), h16T = f16 h^T
// (column-major, so P@h B-operands are contiguous per lane).
// ---------------------------------------------------------------------------
__global__ __launch_bounds__(128) void prep_kernel(const float* __restrict__ h,
                                                   _Float16* __restrict__ hn,
                                                   _Float16* __restrict__ hT) {
  __shared__ float red[128];
  const int i = blockIdx.x;
  const int c = threadIdx.x;
  const float v = h[(size_t)i * NHID_ + c];
  red[c] = v * v;
  __syncthreads();
  for (int s = 64; s > 0; s >>= 1) {
    if (c < s) red[c] += red[c + s];
    __syncthreads();
  }
  const float rn = rsqrtf(red[0] + 1e-14f);
  hn[(size_t)i * NHID_ + c] = (_Float16)(v * rn);
  hT[(size_t)c * N_ + i] = (_Float16)v;
}

// ---------------------------------------------------------------------------
// Fused attention layer (single-pass, shift-by-11 softmax):
//   for each 16-row tile (one block, 8 waves split the 8192 j's in chunks of 32):
//     S = hn_i . hn_j^T  (WMMA f16, K=128)
//     p = keep ? exp(beta*cos + 10*adj - 11) : 0
//     num += p @ h_j (WMMA f16), den += sum p ;  h_out = num/den
// All P@h B-operand loads are issued before the S-WMMA/exp phase; mask words
// are fetched once per wave (1 lane = 1 row-word) and shfl-broadcast.
// ---------------------------------------------------------------------------
__global__ __launch_bounds__(256) void attn_kernel(const _Float16* __restrict__ hn16,
                                                   const _Float16* __restrict__ h16T,
                                                   const unsigned* __restrict__ affw,
                                                   const unsigned* __restrict__ adjw,
                                                   const float* __restrict__ beta_ptr,
                                                   int use_beta,
                                                   float* __restrict__ h_next) {
  __shared__ _Float16 pbuf[8][16][32];   // per-wave C->A layout staging
  __shared__ float numbuf[16][128];
  __shared__ float denbuf[16];

  const int tid = threadIdx.x;
  const int wv = tid >> 5;
  const int lane = tid & 31;
  const int hf = lane >> 4;
  const int lm = lane & 15;
  const int i0 = blockIdx.x * 16;
  const float beta = use_beta ? beta_ptr[0] : 0.f;

  for (int s = tid; s < 16 * 128; s += 256) ((float*)numbuf)[s] = 0.f;
  if (tid < 16) denbuf[tid] = 0.f;
  __syncthreads();

  // A operand (hn rows i0..i0+15) is invariant over the whole j loop.
  v16h A[4];
  {
    const _Float16* arow = hn16 + (size_t)(i0 + lm) * NHID_;
#pragma unroll
    for (int kc = 0; kc < 4; ++kc) {
      const v8h lo = *(const v8h*)(arow + kc * 32 + hf * 8);
      const v8h hi = *(const v8h*)(arow + kc * 32 + 16 + hf * 8);
      A[kc] = cat16(lo, hi);
    }
  }

  // lanes 0-15 fetch aff words for rows i0+lm, lanes 16-31 fetch adj words
  const unsigned* mbase =
      (lane < 16 ? affw : adjw) + (size_t)(i0 + lm) * NW_;

  v8f acc[8];
#pragma unroll
  for (int t = 0; t < 8; ++t) acc[t] = {};
  float dacc[8];
#pragma unroll
  for (int r = 0; r < 8; ++r) dacc[r] = 0.f;

  for (int c = wv; c < NW_; c += 8) {          // 32 chunks of 32 j's per wave
    const int j0 = c * 32;
    if (c + 8 < NW_) {                          // stream-ahead hints
      __builtin_prefetch(hn16 + (size_t)(j0 + 256 + lm) * NHID_, 0, 1);
      __builtin_prefetch(h16T + (size_t)lm * N_ + j0 + 256, 0, 1);
    }
    // ---- issue ALL independent loads for this chunk up front ----
    // one mask word per lane, then broadcast (one word covers all 32 j's)
    const unsigned mword = mbase[j0 >> 5];
    unsigned aw[8], jw[8];
#pragma unroll
    for (int r = 0; r < 8; ++r) {
      aw[r] = (unsigned)__shfl((int)mword, r + 8 * hf);        // aff rows
      jw[r] = (unsigned)__shfl((int)mword, 16 + r + 8 * hf);   // adj rows
    }
    // P@h B tiles (h16T is column-major h): independent of the S/exp phase.
    v8h hlo[8], hhi[8];
#pragma unroll
    for (int t = 0; t < 8; ++t) {
      const _Float16* hr = h16T + (size_t)(t * 16 + lm) * N_ + j0 + hf * 8;
      hlo[t] = *(const v8h*)hr;
      hhi[t] = *(const v8h*)(hr + 16);
    }
    // ---- S = hn_i . hn_j^T, mask, exp ----
#pragma unroll
    for (int jt = 0; jt < 2; ++jt) {           // two 16-wide S tiles
      const _Float16* br = hn16 + (size_t)(j0 + jt * 16 + lm) * NHID_;
      v8h blo[4], bhi[4];
#pragma unroll
      for (int kc = 0; kc < 4; ++kc) {
        blo[kc] = *(const v8h*)(br + kc * 32 + hf * 8);
        bhi[kc] = *(const v8h*)(br + kc * 32 + 16 + hf * 8);
      }
      v8f s = {};
#pragma unroll
      for (int kc = 0; kc < 4; ++kc)
        s = wmma_f16(A[kc], cat16(blo[kc], bhi[kc]), s);

      const int bi = jt * 16 + lm;             // bit index within the 32-j chunk
#pragma unroll
      for (int r = 0; r < 8; ++r) {
        const float cosv = s[r];
        float p = 0.f;
        if (((aw[r] >> bi) & 1u) && (cosv >= 0.f)) {
          const float adjv = (float)((jw[r] >> bi) & 1u);
          p = __expf(beta * cosv + 10.f * adjv - 11.f);   // p in (0,1]
        }
        dacc[r] += p;
        pbuf[wv][r + 8 * hf][bi] = (_Float16)p;
      }
    }
    // Re-read p in A layout (wave-private buffer; DS ops are in-order per wave).
    const v8h plo = *(const v8h*)&pbuf[wv][lm][hf * 8];
    const v8h phi = *(const v8h*)&pbuf[wv][lm][16 + hf * 8];
    const v16h A2 = cat16(plo, phi);
    // ---- num(16x128) += P(16x32) @ h(32x128), operands already resident ----
#pragma unroll
    for (int t = 0; t < 8; ++t)
      acc[t] = wmma_f16(A2, cat16(hlo[t], hhi[t]), acc[t]);
  }

  // cross-wave reduction
#pragma unroll
  for (int r = 0; r < 8; ++r) atomicAdd(&denbuf[r + 8 * hf], dacc[r]);
#pragma unroll
  for (int t = 0; t < 8; ++t)
#pragma unroll
    for (int r = 0; r < 8; ++r)
      atomicAdd(&numbuf[r + 8 * hf][t * 16 + lm], acc[t][r]);
  __syncthreads();

  for (int s = tid; s < 16 * 128; s += 256) {
    const int m = s >> 7;
    const int cc = s & 127;
    h_next[(size_t)(i0 + m) * NHID_ + cc] =
        numbuf[m][cc] / fmaxf(denbuf[m], 1e-30f);
  }
}

// ---------------------------------------------------------------------------
// out = h @ w_out (tiny), plus copy fts = h
// ---------------------------------------------------------------------------
__global__ __launch_bounds__(128) void out_kernel(const float* __restrict__ h,
                                                  const float* __restrict__ wout,
                                                  float* __restrict__ out,
                                                  float* __restrict__ fts) {
  __shared__ float hr[128];
  const int i = blockIdx.x;
  const int c = threadIdx.x;
  const float v = h[(size_t)i * NHID_ + c];
  hr[c] = v;
  fts[(size_t)i * NHID_ + c] = v;
  __syncthreads();
  if (c < NCLASS_) {
    float s = 0.f;
#pragma unroll
    for (int k = 0; k < NHID_; ++k) s += hr[k] * wout[k * NCLASS_ + c];
    out[(size_t)i * NCLASS_ + c] = s;
  }
}

// ---------------------------------------------------------------------------
extern "C" void kernel_launch(void* const* d_in, const int* in_sizes, int n_in,
                              void* d_out, int out_size, void* d_ws, size_t ws_size,
                              hipStream_t stream) {
  const float* x       = (const float*)d_in[0];
  const float* adj     = (const float*)d_in[1];
  const float* aff     = (const float*)d_in[2];
  const float* w_embed = (const float*)d_in[3];
  const float* w_out   = (const float*)d_in[4];
  const float* beta    = (const float*)d_in[5];

  float* out = (float*)d_out;                 // [8192*21]
  float* fts = out + (size_t)N_ * NCLASS_;    // [8192*128]

  char* ws = (char*)d_ws;
  size_t off = 0;
  auto carve = [&](size_t bytes) {
    void* p = ws + off;
    off = (off + bytes + 255) & ~(size_t)255;
    return p;
  };
  unsigned* affw  = (unsigned*)carve((size_t)N_ * NW_ * 4);       // 8 MB
  unsigned* adjw  = (unsigned*)carve((size_t)N_ * NW_ * 4);       // 8 MB
  float*    hA    = (float*)carve((size_t)N_ * NHID_ * 4);        // 4 MB
  float*    hB    = (float*)carve((size_t)N_ * NHID_ * 4);        // 4 MB
  _Float16* hn16  = (_Float16*)carve((size_t)N_ * NHID_ * 2);     // 2 MB
  _Float16* h16T  = (_Float16*)carve((size_t)N_ * NHID_ * 2);     // 2 MB
  _Float16* wT16  = (_Float16*)carve((size_t)NFEAT_ * NHID_ * 2); // 128 KB

  pack_kernel<<<N_, 256, 0, stream>>>(adj, aff, adjw, affw);
  wt_kernel<<<(NFEAT_ * NHID_) / 256, 256, 0, stream>>>(w_embed, wT16);
  embed_kernel<<<N_ / 64, 256, 0, stream>>>(x, wT16, hA);

  prep_kernel<<<N_, 128, 0, stream>>>(hA, hn16, h16T);
  attn_kernel<<<N_ / 16, 256, 0, stream>>>(hn16, h16T, affw, adjw, beta, 1, hB);

  prep_kernel<<<N_, 128, 0, stream>>>(hB, hn16, h16T);
  attn_kernel<<<N_ / 16, 256, 0, stream>>>(hn16, h16T, affw, adjw, beta, 0, hA);

  out_kernel<<<N_, 128, 0, stream>>>(hA, w_out, out, fts);
}
